// HeteroGNN_80625126080956
// MI455X (gfx1250) — compile-verified, matrix operand order checked by reading, not run on hardware
//
#include <hip/hip_runtime.h>

typedef __attribute__((ext_vector_type(16))) _Float16 v16h;
typedef __attribute__((ext_vector_type(8)))  _Float16 v8h;
typedef __attribute__((ext_vector_type(4)))  _Float16 v4h;
typedef __attribute__((ext_vector_type(8)))  float    v8f;

#define HIDDEN 128

__device__ __forceinline__ void atomAddF(float* p, float v) {
  __hip_atomic_fetch_add(p, v, __ATOMIC_RELAXED, __HIP_MEMORY_SCOPE_AGENT);
}

// ---- degree counting + inverse-sqrt normalization --------------------------
__global__ void k_deg(const int* __restrict__ src, const int* __restrict__ dst,
                      float* __restrict__ degS, float* __restrict__ degD, int E) {
  int e = blockIdx.x * blockDim.x + threadIdx.x;
  if (e < E) {
    atomAddF(&degS[src[e]], 1.0f);
    atomAddF(&degD[dst[e]], 1.0f);
  }
}

__global__ void k_invsqrt(float* __restrict__ p, int n) {
  int i = blockIdx.x * blockDim.x + threadIdx.x;
  if (i < n) { float d = p[i]; p[i] = d > 0.f ? rsqrtf(d) : 0.f; }
}

// ---- fp32 -> fp16 convert --------------------------------------------------
__global__ void k_f32tof16(const float* __restrict__ s, _Float16* __restrict__ d, int n) {
  int i = blockIdx.x * blockDim.x + threadIdx.x;
  if (i < n) d[i] = (_Float16)s[i];
}

// W [K x N] row-major  ->  Wt [N x Kp] row-major fp16 (zero pad K..Kp)
__global__ void k_wtrans(const float* __restrict__ W, _Float16* __restrict__ Wt,
                         int K, int N, int Kp) {
  int i = blockIdx.x * blockDim.x + threadIdx.x;
  if (i < N * Kp) {
    int n = i / Kp, k = i % Kp;
    Wt[i] = (k < K) ? (_Float16)W[(size_t)k * N + n] : (_Float16)0.f;
  }
}

// ---- WMMA GEMM: C[M x N] = A[M x K] * Bt[N x K]^T (+bias, opt relu), fp16 out
// Bt staged once per block into LDS (row stride K+8 halves -> bank-conflict free)
__device__ __forceinline__ v16h frag16(v8h lo, v8h hi) {
  return __builtin_shufflevector(lo, hi, 0,1,2,3,4,5,6,7,8,9,10,11,12,13,14,15);
}

template<int NT, int K>
__global__ void k_gemm(const _Float16* __restrict__ A, const _Float16* __restrict__ Bt,
                       const float* __restrict__ bias, _Float16* __restrict__ C,
                       int Mtiles, int relu) {
  constexpr int LDB = K + 8;                    // halves; 272B stride = 68 dwords
  __shared__ _Float16 sB[NT * 16 * LDB];

  // cooperative stage of Bt (NT*16 x K) into LDS, 16B chunks
  constexpr int CH = K / 8;
  for (int idx = threadIdx.x; idx < NT * 16 * CH; idx += 256) {
    int row = idx / CH, c = idx - row * CH;
    *(v8h*)&sB[row * LDB + c * 8] = *(const v8h*)&Bt[(size_t)row * K + c * 8];
  }
  __syncthreads();

  int lane = threadIdx.x & 31;
  int mt = blockIdx.x * 8 + (threadIdx.x >> 5);
  if (mt >= Mtiles) return;
  int m0 = mt << 4;
  int r  = lane & 15;
  int kh = (lane >> 4) << 3;          // lanes 16-31 start at K+8 per ISA layout
  const _Float16* arow  = A + (size_t)(m0 + r) * K + kh;
  const _Float16* bbase = &sB[r * LDB + kh];

  v8f zero = {0.f,0.f,0.f,0.f,0.f,0.f,0.f,0.f};
  v8f acc[NT];
#pragma unroll
  for (int t = 0; t < NT; ++t) acc[t] = zero;

#pragma unroll
  for (int k = 0; k < K; k += 32) {
    v16h a = frag16(*(const v8h*)(arow + k), *(const v8h*)(arow + k + 16));
#pragma unroll
    for (int t = 0; t < NT; ++t) {
      const _Float16* bp = bbase + t * 16 * LDB + k;
      v16h b = frag16(*(const v8h*)bp, *(const v8h*)(bp + 16));
      acc[t] = __builtin_amdgcn_wmma_f32_16x16x32_f16(
                 false, a, false, b, (short)0, acc[t], false, false);
    }
  }

  constexpr int N = NT * 16;
  int col   = lane & 15;
  int rbase = m0 + ((lane >> 4) << 3);   // lane>=16 holds rows 8..15 of tile
#pragma unroll
  for (int t = 0; t < NT; ++t) {
    float bv = bias ? bias[t * 16 + col] : 0.f;
#pragma unroll
    for (int j = 0; j < 8; ++j) {
      float v = acc[t][j] + bv;
      if (relu) v = fmaxf(v, 0.f);
      C[(size_t)(rbase + j) * N + t * 16 + col] = (_Float16)v;
    }
  }
}

// ---- edge scatter: out[dst] += scale * ns[src]*nd[dst] * H[src] ------------
// one wave per edge, 4 channels per lane (128 channels)
__global__ void k_scatter(const int* __restrict__ src, const int* __restrict__ dst,
                          const float* __restrict__ ns, const float* __restrict__ nd,
                          const _Float16* __restrict__ H, float* __restrict__ out,
                          float scale, int E) {
  int tid  = blockIdx.x * blockDim.x + threadIdx.x;
  int e    = tid >> 5;
  int lane = tid & 31;
  if (e >= E) return;
  int s = src[e], d = dst[e];
  float w = ns[s] * nd[d] * scale;
  v4h h4 = *(const v4h*)(H + (size_t)s * HIDDEN + lane * 4);
  float* op = out + (size_t)d * HIDDEN + lane * 4;
#pragma unroll
  for (int i = 0; i < 4; ++i) atomAddF(&op[i], (float)h4[i] * w);
}

// ---- finalize: relu(acc + bias) -> fp16 ------------------------------------
__global__ void k_fin_a(const float* __restrict__ acc, const float* __restrict__ b,
                        _Float16* __restrict__ out, int n) {
  int i = blockIdx.x * blockDim.x + threadIdx.x;
  if (i < n) {
    float v = acc[i] + b[i & (HIDDEN - 1)];
    out[i] = (_Float16)fmaxf(v, 0.f);
  }
}

__global__ void k_fin_t(const float* __restrict__ acc, const float* __restrict__ b1,
                        const float* __restrict__ b2, _Float16* __restrict__ out, int n) {
  int i = blockIdx.x * blockDim.x + threadIdx.x;
  if (i < n) {
    int c = i & (HIDDEN - 1);
    float v = acc[i] + 0.5f * (b1[c] + b2[c]);
    out[i] = (_Float16)fmaxf(v, 0.f);
  }
}

// ---- final 64->2 layer + log_softmax ---------------------------------------
__global__ void k_cls(const _Float16* __restrict__ H, const float* __restrict__ W2,
                      const float* __restrict__ b2, float* __restrict__ out, int M) {
  int r = blockIdx.x * blockDim.x + threadIdx.x;
  if (r >= M) return;
  const _Float16* h = H + (size_t)r * 64;
  float l0 = b2[0], l1 = b2[1];
#pragma unroll 16
  for (int j = 0; j < 64; ++j) {
    float x = (float)h[j];
    l0 += x * W2[2 * j];
    l1 += x * W2[2 * j + 1];
  }
  float mx  = fmaxf(l0, l1);
  float lse = mx + logf(expf(l0 - mx) + expf(l1 - mx));
  out[2 * r]     = l0 - lse;
  out[2 * r + 1] = l1 - lse;
}

extern "C" void kernel_launch(void* const* d_in, const int* in_sizes, int n_in,
                              void* d_out, int out_size, void* d_ws, size_t ws_size,
                              hipStream_t stream) {
  (void)n_in; (void)out_size; (void)ws_size;

  // setup_inputs() insertion order (params flattened recursively):
  const float* x_user = (const float*)d_in[0];
  const int*   e_owns = (const int*)d_in[3];
  const int*   e_init = (const int*)d_in[4];
  const int*   e_recv = (const int*)d_in[5];
  const float* W_user = (const float*)d_in[6];
  const float* b_user = (const float*)d_in[7];
  // convs[l] at 12 + 6*l : owns_W, owns_b, init_W, init_b, recv_W, recv_b
  const float* owns_W = (const float*)d_in[18];  // convs[1].owns_W (feeds final a)
  const float* owns_b = (const float*)d_in[19];
  const float* init_W = (const float*)d_in[26];  // convs[2]
  const float* init_b = (const float*)d_in[27];
  const float* recv_W = (const float*)d_in[28];
  const float* recv_b = (const float*)d_in[29];
  const float* cls_W1 = (const float*)d_in[30];
  const float* cls_b1 = (const float*)d_in[31];
  const float* cls_W2 = (const float*)d_in[32];
  const float* cls_b2 = (const float*)d_in[33];

  const int NU  = in_sizes[0] / 32;   // users
  const int NA  = in_sizes[1] / 48;   // accounts
  const int NTr = in_sizes[2] / 64;   // transactions
  const int EO  = in_sizes[3] / 2;
  const int EI  = in_sizes[4] / 2;
  const int ER  = in_sizes[5] / 2;

  size_t off = 0;
  auto alloc = [&](size_t bytes) -> void* {
    off = (off + 255) & ~(size_t)255;
    void* p = (char*)d_ws + off;
    off += bytes;
    return p;
  };

  float* dis_u  = (float*)alloc(sizeof(float) * NU);
  float* did_a  = (float*)alloc(sizeof(float) * NA);
  float* dis_ai = (float*)alloc(sizeof(float) * NA);
  float* did_ti = (float*)alloc(sizeof(float) * NTr);
  float* dis_ar = (float*)alloc(sizeof(float) * NA);
  float* did_tr = (float*)alloc(sizeof(float) * NTr);

  _Float16* xu_h  = (_Float16*)alloc(sizeof(_Float16) * (size_t)NU * 32);
  _Float16* WuT   = (_Float16*)alloc(sizeof(_Float16) * 128 * 32);
  _Float16* WoT   = (_Float16*)alloc(sizeof(_Float16) * 128 * 128);
  _Float16* WiT   = (_Float16*)alloc(sizeof(_Float16) * 128 * 128);
  _Float16* WrT   = (_Float16*)alloc(sizeof(_Float16) * 128 * 128);
  _Float16* Wc1T  = (_Float16*)alloc(sizeof(_Float16) * 64 * 128);
  _Float16* u_h   = (_Float16*)alloc(sizeof(_Float16) * (size_t)NU * HIDDEN);
  _Float16* h_msg = (_Float16*)alloc(sizeof(_Float16) * (size_t)NA * HIDDEN);
  float*    a_acc = (float*)   alloc(sizeof(float)    * (size_t)NA * HIDDEN);
  _Float16* a_h   = (_Float16*)alloc(sizeof(_Float16) * (size_t)NA * HIDDEN);
  float*    t_acc = (float*)   alloc(sizeof(float)    * (size_t)NTr * HIDDEN);
  _Float16* t_h   = (_Float16*)alloc(sizeof(_Float16) * (size_t)NTr * HIDDEN);
  _Float16* hcls  = (_Float16*)a_acc;  // a_acc is dead by the time hcls is written

  const int B = 256;
  auto cdiv = [](long a, long b) -> int { return (int)((a + b - 1) / b); };

  // 1) degree norms
  hipMemsetAsync(dis_u,  0, sizeof(float) * NU,  stream);
  hipMemsetAsync(did_a,  0, sizeof(float) * NA,  stream);
  hipMemsetAsync(dis_ai, 0, sizeof(float) * NA,  stream);
  hipMemsetAsync(did_ti, 0, sizeof(float) * NTr, stream);
  hipMemsetAsync(dis_ar, 0, sizeof(float) * NA,  stream);
  hipMemsetAsync(did_tr, 0, sizeof(float) * NTr, stream);
  k_deg<<<cdiv(EO, B), B, 0, stream>>>(e_owns, e_owns + EO, dis_u,  did_a,  EO);
  k_deg<<<cdiv(EI, B), B, 0, stream>>>(e_init, e_init + EI, dis_ai, did_ti, EI);
  k_deg<<<cdiv(ER, B), B, 0, stream>>>(e_recv, e_recv + ER, dis_ar, did_tr, ER);
  k_invsqrt<<<cdiv(NU,  B), B, 0, stream>>>(dis_u,  NU);
  k_invsqrt<<<cdiv(NA,  B), B, 0, stream>>>(did_a,  NA);
  k_invsqrt<<<cdiv(NA,  B), B, 0, stream>>>(dis_ai, NA);
  k_invsqrt<<<cdiv(NTr, B), B, 0, stream>>>(did_ti, NTr);
  k_invsqrt<<<cdiv(NA,  B), B, 0, stream>>>(dis_ar, NA);
  k_invsqrt<<<cdiv(NTr, B), B, 0, stream>>>(did_tr, NTr);

  // 2) fp16 conversions / weight transposes
  k_f32tof16<<<cdiv((size_t)NU * 32, B), B, 0, stream>>>(x_user, xu_h, NU * 32);
  k_wtrans<<<cdiv(128 * 32,  B), B, 0, stream>>>(W_user, WuT,  32, 128, 32);
  k_wtrans<<<cdiv(128 * 128, B), B, 0, stream>>>(owns_W, WoT, 128, 128, 128);
  k_wtrans<<<cdiv(128 * 128, B), B, 0, stream>>>(init_W, WiT, 128, 128, 128);
  k_wtrans<<<cdiv(128 * 128, B), B, 0, stream>>>(recv_W, WrT, 128, 128, 128);
  k_wtrans<<<cdiv(64 * 128,  B), B, 0, stream>>>(cls_W1, Wc1T, 128, 64, 128);

  // 3) u = x_user @ W_user + b_user  (fp16 out)
  int MtU = NU / 16, MtA = NA / 16, MtT = NTr / 16;
  k_gemm<8, 32><<<cdiv(MtU, 8), 256, 0, stream>>>(xu_h, WuT, b_user, u_h, MtU, 0);

  // 4) a = relu(gcn(u, owns, convs[1]))
  k_gemm<8, 128><<<cdiv(MtU, 8), 256, 0, stream>>>(u_h, WoT, nullptr, h_msg, MtU, 0);
  hipMemsetAsync(a_acc, 0, sizeof(float) * (size_t)NA * HIDDEN, stream);
  k_scatter<<<cdiv((size_t)EO * 32, B), B, 0, stream>>>(
      e_owns, e_owns + EO, dis_u, did_a, h_msg, a_acc, 1.0f, EO);
  k_fin_a<<<cdiv((size_t)NA * HIDDEN, B), B, 0, stream>>>(a_acc, owns_b, a_h, NA * HIDDEN);

  // 5) t = relu(0.5*(gcn(a, init, convs[2]) + gcn(a, recv, convs[2])))
  hipMemsetAsync(t_acc, 0, sizeof(float) * (size_t)NTr * HIDDEN, stream);
  k_gemm<8, 128><<<cdiv(MtA, 8), 256, 0, stream>>>(a_h, WiT, nullptr, h_msg, MtA, 0);
  k_scatter<<<cdiv((size_t)EI * 32, B), B, 0, stream>>>(
      e_init, e_init + EI, dis_ai, did_ti, h_msg, t_acc, 0.5f, EI);
  k_gemm<8, 128><<<cdiv(MtA, 8), 256, 0, stream>>>(a_h, WrT, nullptr, h_msg, MtA, 0);
  k_scatter<<<cdiv((size_t)ER * 32, B), B, 0, stream>>>(
      e_recv, e_recv + ER, dis_ar, did_tr, h_msg, t_acc, 0.5f, ER);
  k_fin_t<<<cdiv((size_t)NTr * HIDDEN, B), B, 0, stream>>>(
      t_acc, init_b, recv_b, t_h, NTr * HIDDEN);

  // 6) classifier: relu(t @ cls_W1 + b1) -> logits -> log_softmax
  k_gemm<4, 128><<<cdiv(MtT, 8), 256, 0, stream>>>(t_h, Wc1T, cls_b1, hcls, MtT, 1);
  k_cls<<<cdiv(NTr, B), B, 0, stream>>>(hcls, cls_W2, cls_b2, (float*)d_out, NTr);
}
